// GenuineLayer_35485019799940
// MI455X (gfx1250) — matrix operand order repaired and checked by reading, not compiled, not run on hardware
//
#include <hip/hip_runtime.h>
#include <hip/hip_bf16.h>

#define DMODEL 1024
#define NHEAD  16
#define HDIM   64
#define BATCH  2
#define SEQ    2048
#define MTOT   (BATCH*SEQ)      // 4096
#define FF     (4*DMODEL)       // 4096
#define LN_EPS 1e-5f

typedef __attribute__((ext_vector_type(16))) __bf16          v16bf;
typedef __attribute__((ext_vector_type(8)))  float           v8f;
typedef __attribute__((ext_vector_type(8)))  unsigned short  v8u16;
typedef __attribute__((ext_vector_type(16))) unsigned short  v16u16;

union Frag16 { v16u16 u; v16bf b; };

__device__ __forceinline__ unsigned short f2bf_u(float f) {
  union { float f; unsigned u; } x; x.f = f;
  unsigned r = x.u + 0x7FFFu + ((x.u >> 16) & 1u);   // round-to-nearest-even
  return (unsigned short)(r >> 16);
}

__device__ __forceinline__ v8f vzero8() {
  v8f z = {0.f,0.f,0.f,0.f,0.f,0.f,0.f,0.f};
  return z;
}

// ---- CDNA5 async global->LDS copy (ASYNCcnt path) -------------------------
// Per-lane: LDS[vdst] <- MEM[vaddr], 16 bytes, tracked by ASYNCcnt.
__device__ __forceinline__ void async_copy_b128(void* lds, const void* gaddr) {
  unsigned lds_off = (unsigned)(unsigned long long)lds;  // flat LDS ptr low 32 = LDS offset
  asm volatile("global_load_async_to_lds_b128 %0, %1, off"
               :: "v"(lds_off), "v"(gaddr) : "memory");
}
__device__ __forceinline__ void async_wait0() {
  asm volatile("s_wait_asynccnt 0x0" ::: "memory");
}

// ---- WMMA fragment loaders (layouts per CDNA5 ISA 7.12.2) -----------------
// A 16x32 bf16, row-major [16][ld]: lanes 0-15 row=lane, K {0..7},{16..23};
// lanes 16-31 row=lane-16, K {8..15},{24..31}  -> two 16B chunks per lane.
__device__ __forceinline__ v16bf load_a_bf16(const unsigned short* __restrict__ tile,
                                             int ld, int lane) {
  int r  = lane & 15;
  int hs = (lane >> 4) & 1;
  const unsigned short* p = tile + (size_t)r * ld + hs * 8;
  v8u16 c0 = *(const v8u16*)(p);
  v8u16 c1 = *(const v8u16*)(p + 16);
  Frag16 f;
#pragma unroll
  for (int i = 0; i < 8; ++i) { f.u[i] = c0[i]; f.u[i + 8] = c1[i]; }
  return f.b;
}

// B 32x16 bf16 from BT stored row-major [N][ld] (row n = output column,
// contiguous K): lane n=lane&15, K {0..7},{16..23} (lanes 0-15) or {8..15},{24..31} (lanes 16-31).
__device__ __forceinline__ v16bf load_b_bf16(const unsigned short* __restrict__ tile,
                                             int ld, int lane) {
  int n  = lane & 15;
  int hs = (lane >> 4) & 1;
  const unsigned short* p = tile + (size_t)n * ld + hs * 8;
  v8u16 c0 = *(const v8u16*)(p);
  v8u16 c1 = *(const v8u16*)(p + 16);
  Frag16 f;
#pragma unroll
  for (int i = 0; i < 8; ++i) { f.u[i] = c0[i]; f.u[i + 8] = c1[i]; }
  return f.b;
}

__device__ __forceinline__ v8f wmma_bf16(v16bf a, v16bf b, v8f c) {
  return __builtin_amdgcn_wmma_f32_16x16x32_bf16(false, a, false, b, (short)0, c,
                                                 false, false);
}

// ---- LayerNorm 1: h = bf16(LN(x)); row staged in LDS via async copy -------
__global__ void __launch_bounds__(256)
ln1_kernel(const float* __restrict__ x, const float* __restrict__ g,
           const float* __restrict__ bt, unsigned short* __restrict__ h) {
  __shared__ float xs[DMODEL];
  __shared__ float red[256];
  int row = blockIdx.x;
  const float* xr = x + (size_t)row * DMODEL;
  // one b128 per lane stages the whole 4KB row
  async_copy_b128(&xs[threadIdx.x * 4], xr + threadIdx.x * 4);
  async_wait0();
  __syncthreads();

  float s = 0.f;
#pragma unroll
  for (int c = 0; c < 4; ++c) s += xs[threadIdx.x * 4 + c];
  red[threadIdx.x] = s; __syncthreads();
  for (int o = 128; o > 0; o >>= 1) {
    if (threadIdx.x < o) red[threadIdx.x] += red[threadIdx.x + o];
    __syncthreads();
  }
  float mean = red[0] / DMODEL;
  __syncthreads();
  float v = 0.f;
#pragma unroll
  for (int c = 0; c < 4; ++c) { float d = xs[threadIdx.x * 4 + c] - mean; v += d * d; }
  red[threadIdx.x] = v; __syncthreads();
  for (int o = 128; o > 0; o >>= 1) {
    if (threadIdx.x < o) red[threadIdx.x] += red[threadIdx.x + o];
    __syncthreads();
  }
  float rstd = rsqrtf(red[0] / DMODEL + LN_EPS);
  unsigned short* hr = h + (size_t)row * DMODEL;
#pragma unroll
  for (int c = 0; c < 4; ++c) {
    int i = threadIdx.x * 4 + c;
    hr[i] = f2bf_u((xs[i] - mean) * rstd * g[i] + bt[i]);
  }
}

// ---- y = x + attn_out ; h2 = bf16(LN(y)); async-staged rows ---------------
__global__ void __launch_bounds__(256)
addln2_kernel(const float* __restrict__ x, const float* __restrict__ a,
              const float* __restrict__ g, const float* __restrict__ bt,
              float* __restrict__ y, unsigned short* __restrict__ h2) {
  __shared__ float xs[DMODEL];
  __shared__ float as[DMODEL];
  __shared__ float ys[DMODEL];
  __shared__ float red[256];
  int row = blockIdx.x;
  const float* xr = x + (size_t)row * DMODEL;
  const float* ar = a + (size_t)row * DMODEL;
  float* yr = y + (size_t)row * DMODEL;
  async_copy_b128(&xs[threadIdx.x * 4], xr + threadIdx.x * 4);
  async_copy_b128(&as[threadIdx.x * 4], ar + threadIdx.x * 4);
  async_wait0();
  __syncthreads();

  float s = 0.f;
#pragma unroll
  for (int c = 0; c < 4; ++c) {
    int i = threadIdx.x * 4 + c;
    float v = xs[i] + as[i];
    ys[i] = v; yr[i] = v; s += v;
  }
  red[threadIdx.x] = s; __syncthreads();
  for (int o = 128; o > 0; o >>= 1) {
    if (threadIdx.x < o) red[threadIdx.x] += red[threadIdx.x + o];
    __syncthreads();
  }
  float mean = red[0] / DMODEL;
  __syncthreads();
  float v = 0.f;
#pragma unroll
  for (int c = 0; c < 4; ++c) { float d = ys[threadIdx.x * 4 + c] - mean; v += d * d; }
  red[threadIdx.x] = v; __syncthreads();
  for (int o = 128; o > 0; o >>= 1) {
    if (threadIdx.x < o) red[threadIdx.x] += red[threadIdx.x + o];
    __syncthreads();
  }
  float rstd = rsqrtf(red[0] / DMODEL + LN_EPS);
  unsigned short* hr = h2 + (size_t)row * DMODEL;
#pragma unroll
  for (int c = 0; c < 4; ++c) {
    int i = threadIdx.x * 4 + c;
    hr[i] = f2bf_u((ys[i] - mean) * rstd * g[i] + bt[i]);
  }
}

// ---- QKV weight transpose+convert: [H,D,HD] f32 -> [H*HD, D] bf16 ---------
__global__ void qkvw_transpose_kernel(const float* __restrict__ Wq,
                                      const float* __restrict__ Wk,
                                      const float* __restrict__ Wv,
                                      unsigned short* __restrict__ WqT,
                                      unsigned short* __restrict__ WkT,
                                      unsigned short* __restrict__ WvT) {
  size_t idx = (size_t)blockIdx.x * blockDim.x + threadIdx.x;
  const size_t total = (size_t)NHEAD * HDIM * DMODEL;
  if (idx >= total) return;
  const float* in = (blockIdx.y == 0) ? Wq : (blockIdx.y == 1) ? Wk : Wv;
  unsigned short* out = (blockIdx.y == 0) ? WqT : (blockIdx.y == 1) ? WkT : WvT;
  int d  = (int)(idx % DMODEL);
  int f  = (int)((idx / DMODEL) % HDIM);
  int hh = (int)(idx / ((size_t)DMODEL * HDIM));
  out[idx] = f2bf_u(in[((size_t)hh * DMODEL + d) * HDIM + f]);
}

// ---- generic transpose+convert: out[n*K+k] = bf16(in[k*N+n]) --------------
__global__ void transpose_bf16_kernel(const float* __restrict__ in,
                                      unsigned short* __restrict__ out,
                                      int K, int N) {
  size_t idx = (size_t)blockIdx.x * blockDim.x + threadIdx.x;
  if (idx >= (size_t)K * N) return;
  int k = (int)(idx % K);
  int n = (int)(idx / K);
  out[idx] = f2bf_u(in[(size_t)k * N + n]);
}

// ---- QKV GEMM: [4096x1024] @ [1024x1024] per matrix, bf16 WMMA ------------
__global__ void __launch_bounds__(256)
qkv_gemm_kernel(const unsigned short* __restrict__ h,
                const unsigned short* __restrict__ WqT,
                const unsigned short* __restrict__ WkT,
                const unsigned short* __restrict__ WvT,
                const float* __restrict__ bq, const float* __restrict__ bk,
                const float* __restrict__ bv,
                unsigned short* __restrict__ q_ws,
                unsigned short* __restrict__ k_ws,
                unsigned short* __restrict__ vT_ws) {
  int wave = (int)((blockIdx.x * blockDim.x + threadIdx.x) >> 5);
  int lane = threadIdx.x & 31;
  int ntile = wave & 15;          // 16 n-tiles of 64 == one head each
  int mtile = (wave >> 4) & 255;  // 256 m-tiles of 16
  int mat   = wave >> 12;         // 0=Q 1=K 2=V
  if (mat >= 3) return;
  const unsigned short* BT = (mat == 0) ? WqT : (mat == 1) ? WkT : WvT;
  const unsigned short* A  = h + (size_t)mtile * 16 * DMODEL;
  v8f acc[4];
#pragma unroll
  for (int s = 0; s < 4; ++s) acc[s] = vzero8();
  for (int kk = 0; kk < DMODEL; kk += 32) {
    v16bf a = load_a_bf16(A + kk, DMODEL, lane);
#pragma unroll
    for (int s = 0; s < 4; ++s) {
      const unsigned short* bt = BT + (size_t)(ntile * 64 + s * 16) * DMODEL + kk;
      __builtin_prefetch(bt + 32, 0, 0);
      v16bf bf = load_b_bf16(bt, DMODEL, lane);
      acc[s] = wmma_bf16(a, bf, acc[s]);
    }
  }
  const float* bias = (mat == 0) ? bq : (mat == 1) ? bk : bv;
  float scale = (mat == 0) ? 0.125f : 1.0f;   // 1/sqrt(64)
  int m0 = mtile * 16;
#pragma unroll
  for (int s = 0; s < 4; ++s) {
    int n  = ntile * 64 + s * 16 + (lane & 15);
    int hh = n >> 6, f = n & 63;
    float bi = bias[hh * HDIM + f];
#pragma unroll
    for (int r = 0; r < 8; ++r) {
      int m = m0 + r + 8 * (lane >> 4);
      int bb = m >> 11, sidx = m & (SEQ - 1);
      unsigned short bfv = f2bf_u((acc[s][r] + bi) * scale);
      if (mat == 2)
        vT_ws[(((size_t)hh * BATCH + bb) * HDIM + f) * SEQ + sidx] = bfv;
      else if (mat == 0)
        q_ws[(((size_t)hh * BATCH + bb) * SEQ + sidx) * HDIM + f] = bfv;
      else
        k_ws[(((size_t)hh * BATCH + bb) * SEQ + sidx) * HDIM + f] = bfv;
    }
  }
}

// ---- Attention: one block per (head, batch, 16-query tile) ----------------
// LDS: 128KB f32 scores + 64KB bf16 probs + 4KB P@V partials (196KB/WGP ok).
// Phase 1: 8 waves compute score tiles via WMMA.
// Phase 2: wave-per-row softmax; probs -> d_out (f32) and LDS (bf16 once).
// Phase 3: all 8 waves do P@V (4 n-tiles x 2 K-halves), LDS combine.
__global__ void __launch_bounds__(256)
attention_kernel(const unsigned short* __restrict__ q_ws,
                 const unsigned short* __restrict__ k_ws,
                 const unsigned short* __restrict__ vT_ws,
                 float* __restrict__ attn_w,    // [H,B,S,S]
                 float* __restrict__ entropy,   // [H,B,S]
                 float* __restrict__ attn_out)  // [B,S,D] ws
{
  extern __shared__ char smraw[];
  float*          sm   = (float*)smraw;                         // [16][SEQ] f32
  unsigned short* pb   = (unsigned short*)(smraw + 131072);     // [16][SEQ] bf16
  float*          part = (float*)(smraw + 131072 + 65536);      // [4][16][16] f32

  int qt = blockIdx.x & 127;
  int bb = (blockIdx.x >> 7) & 1;
  int hh = blockIdx.x >> 8;
  int lane = threadIdx.x & 31;
  int w = threadIdx.x >> 5;

  const unsigned short* q  = q_ws  + (((size_t)hh * BATCH + bb) * SEQ + qt * 16) * HDIM;
  const unsigned short* kb = k_ws  + ((size_t)hh * BATCH + bb) * SEQ * HDIM;
  const unsigned short* vT = vT_ws + ((size_t)hh * BATCH + bb) * HDIM * SEQ;

  v16bf qa0 = load_a_bf16(q,      HDIM, lane);
  v16bf qa1 = load_a_bf16(q + 32, HDIM, lane);

  // ---- scores ----
  for (int t = w; t < SEQ / 16; t += 8) {
    const unsigned short* kt = kb + (size_t)t * 16 * HDIM;
    v16bf b0 = load_b_bf16(kt,      HDIM, lane);
    v16bf b1 = load_b_bf16(kt + 32, HDIM, lane);
    v8f acc = vzero8();
    acc = wmma_bf16(qa0, b0, acc);
    acc = wmma_bf16(qa1, b1, acc);
    int n  = lane & 15;
    int mb = 8 * (lane >> 4);
#pragma unroll
    for (int r = 0; r < 8; ++r)
      sm[(size_t)(mb + r) * SEQ + t * 16 + n] = acc[r];
  }
  __syncthreads();

  // ---- softmax + entropy: wave w handles rows 2w, 2w+1 ----
#pragma unroll
  for (int rr = 0; rr < 2; ++rr) {
    int row = w * 2 + rr;
    float* srow = sm + (size_t)row * SEQ;
    unsigned short* prow = pb + (size_t)row * SEQ;
    float mx = -1e30f;
    for (int j = lane; j < SEQ; j += 32) mx = fmaxf(mx, srow[j]);
    for (int o = 16; o > 0; o >>= 1) mx = fmaxf(mx, __shfl_xor(mx, o, 32));
    float sum = 0.f;
    for (int j = lane; j < SEQ; j += 32) {
      float e = __expf(srow[j] - mx);
      srow[j] = e; sum += e;
    }
    for (int o = 16; o > 0; o >>= 1) sum += __shfl_xor(sum, o, 32);
    float inv = 1.0f / sum;
    float ent = 0.f;
    float* arow = attn_w + (((size_t)hh * BATCH + bb) * SEQ + qt * 16 + row) * SEQ;
    for (int j = lane; j < SEQ; j += 32) {
      float p = srow[j] * inv;
      prow[j] = f2bf_u(p);        // bf16 once, for P@V
      arow[j] = p;                // f32 output
      ent += p * __log2f(p + 1e-9f);
    }
    for (int o = 16; o > 0; o >>= 1) ent += __shfl_xor(ent, o, 32);
    if (lane == 0)
      entropy[((size_t)hh * BATCH + bb) * SEQ + qt * 16 + row] = -ent;
  }
  __syncthreads();

  // ---- P @ V: 8 waves = 4 n-tiles x 2 K-halves ----
  {
    int nt = w & 3;               // feature tile (16 wide)
    int kh = w >> 2;              // K half: keys [kh*1024, kh*1024+1024)
    v8f acc = vzero8();
    const unsigned short* vt = vT + (size_t)(nt * 16) * SEQ + kh * (SEQ / 2);
    const unsigned short* pk = pb + kh * (SEQ / 2);
    for (int kk = 0; kk < SEQ / 2; kk += 32) {
      v16bf a  = load_a_bf16(pk + kk, SEQ, lane);   // ds_load from LDS probs
      v16bf bf = load_b_bf16(vt + kk, SEQ, lane);
      acc = wmma_bf16(a, bf, acc);
    }
    int n = lane & 15;
    if (kh == 1) {
#pragma unroll
      for (int r = 0; r < 8; ++r) {
        int m = r + 8 * (lane >> 4);
        part[(size_t)nt * 256 + m * 16 + n] = acc[r];
      }
    }
    __syncthreads();
    if (kh == 0) {
#pragma unroll
      for (int r = 0; r < 8; ++r) {
        int m = r + 8 * (lane >> 4);
        float v = acc[r] + part[(size_t)nt * 256 + m * 16 + n];
        attn_out[((size_t)bb * SEQ + qt * 16 + m) * DMODEL + hh * HDIM + nt * 16 + n] = v;
      }
    }
  }
}

// ---- MLP GEMM 1: [4096x1024] @ [1024x4096] + b1, exact GELU, bf16 out -----
__global__ void __launch_bounds__(256)
mlp1_kernel(const unsigned short* __restrict__ h2,
            const unsigned short* __restrict__ W1T,   // [4096][1024] bf16
            const float* __restrict__ b1,
            unsigned short* __restrict__ mid) {       // [4096][4096] bf16
  int wave = (int)((blockIdx.x * blockDim.x + threadIdx.x) >> 5);
  int lane = threadIdx.x & 31;
  int ntile = wave & 63;     // 64 n-tiles of 64
  int mtile = wave >> 6;     // 256 m-tiles of 16
  const unsigned short* A = h2 + (size_t)mtile * 16 * DMODEL;
  v8f acc[4];
#pragma unroll
  for (int s = 0; s < 4; ++s) acc[s] = vzero8();
  for (int kk = 0; kk < DMODEL; kk += 32) {
    v16bf a = load_a_bf16(A + kk, DMODEL, lane);
#pragma unroll
    for (int s = 0; s < 4; ++s) {
      const unsigned short* bt = W1T + (size_t)(ntile * 64 + s * 16) * DMODEL + kk;
      __builtin_prefetch(bt + 32, 0, 0);
      v16bf bf = load_b_bf16(bt, DMODEL, lane);
      acc[s] = wmma_bf16(a, bf, acc[s]);
    }
  }
  int m0 = mtile * 16;
#pragma unroll
  for (int s = 0; s < 4; ++s) {
    int n = ntile * 64 + s * 16 + (lane & 15);
    float bi = b1[n];
#pragma unroll
    for (int r = 0; r < 8; ++r) {
      int m = m0 + r + 8 * (lane >> 4);
      float xg = acc[s][r] + bi;
      float gl = 0.5f * xg * (1.0f + erff(xg * 0.70710678118f));
      mid[(size_t)m * FF + n] = f2bf_u(gl);
    }
  }
}

// ---- MLP GEMM 2: [4096x4096] @ [4096x1024] + b2 + residual y -> out -------
__global__ void __launch_bounds__(256)
mlp2_kernel(const unsigned short* __restrict__ mid,
            const unsigned short* __restrict__ W2T,   // [1024][4096] bf16
            const float* __restrict__ b2,
            const float* __restrict__ y,
            float* __restrict__ out) {
  int wave = (int)((blockIdx.x * blockDim.x + threadIdx.x) >> 5);
  int lane = threadIdx.x & 31;
  int ntile = wave & 15;     // 16 n-tiles of 64
  int mtile = wave >> 4;     // 256 m-tiles of 16
  const unsigned short* A = mid + (size_t)mtile * 16 * FF;
  v8f acc[4];
#pragma unroll
  for (int s = 0; s < 4; ++s) acc[s] = vzero8();
  for (int kk = 0; kk < FF; kk += 32) {
    v16bf a = load_a_bf16(A + kk, FF, lane);
#pragma unroll
    for (int s = 0; s < 4; ++s) {
      const unsigned short* bt = W2T + (size_t)(ntile * 64 + s * 16) * FF + kk;
      __builtin_prefetch(bt + 32, 0, 0);
      v16bf bf = load_b_bf16(bt, FF, lane);
      acc[s] = wmma_bf16(a, bf, acc[s]);
    }
  }
  int m0 = mtile * 16;
#pragma unroll
  for (int s = 0; s < 4; ++s) {
    int n = ntile * 64 + s * 16 + (lane & 15);
    float bi = b2[n];
#pragma unroll
    for (int r = 0; r < 8; ++r) {
      int m = m0 + r + 8 * (lane >> 4);
      out[(size_t)m * DMODEL + n] = y[(size_t)m * DMODEL + n] + acc[s][r] + bi;
    }
  }
}

// ---------------------------------------------------------------------------
extern "C" void kernel_launch(void* const* d_in, const int* in_sizes, int n_in,
                              void* d_out, int out_size, void* d_ws, size_t ws_size,
                              hipStream_t stream) {
  const float* x    = (const float*)d_in[0];
  const float* Wq   = (const float*)d_in[1];
  const float* bq   = (const float*)d_in[2];
  const float* Wk   = (const float*)d_in[3];
  const float* bk   = (const float*)d_in[4];
  const float* Wv   = (const float*)d_in[5];
  const float* bv   = (const float*)d_in[6];
  const float* ln1g = (const float*)d_in[7];
  const float* ln1b = (const float*)d_in[8];
  const float* ln2g = (const float*)d_in[9];
  const float* ln2b = (const float*)d_in[10];
  const float* W1   = (const float*)d_in[11];
  const float* b1   = (const float*)d_in[12];
  const float* W2   = (const float*)d_in[13];
  const float* b2   = (const float*)d_in[14];

  float* out_p   = (float*)d_out;                                  // [B,S,D]
  float* attn_p  = out_p + (size_t)BATCH * SEQ * DMODEL;           // [H,B,S,S]
  float* ent_p   = attn_p + (size_t)NHEAD * BATCH * SEQ * SEQ;     // [H,B,S]

  char* wp = (char*)d_ws;
  auto alloc = [&](size_t bytes) -> char* {
    char* r = wp; wp += (bytes + 255) & ~(size_t)255; return r;
  };
  unsigned short* h_bf  = (unsigned short*)alloc((size_t)MTOT * DMODEL * 2);
  unsigned short* WqT   = (unsigned short*)alloc((size_t)DMODEL * DMODEL * 2);
  unsigned short* WkT   = (unsigned short*)alloc((size_t)DMODEL * DMODEL * 2);
  unsigned short* WvT   = (unsigned short*)alloc((size_t)DMODEL * DMODEL * 2);
  unsigned short* W1T   = (unsigned short*)alloc((size_t)FF * DMODEL * 2);
  unsigned short* W2T   = (unsigned short*)alloc((size_t)DMODEL * FF * 2);
  unsigned short* q_ws  = (unsigned short*)alloc((size_t)NHEAD * BATCH * SEQ * HDIM * 2);
  unsigned short* k_ws  = (unsigned short*)alloc((size_t)NHEAD * BATCH * SEQ * HDIM * 2);
  unsigned short* vT_ws = (unsigned short*)alloc((size_t)NHEAD * BATCH * HDIM * SEQ * 2);
  float*          aout  = (float*)alloc((size_t)MTOT * DMODEL * 4);
  float*          y_ws  = (float*)alloc((size_t)MTOT * DMODEL * 4);
  unsigned short* h2_bf = (unsigned short*)alloc((size_t)MTOT * DMODEL * 2);
  unsigned short* mid   = (unsigned short*)alloc((size_t)MTOT * FF * 2);
  (void)ws_size; (void)in_sizes; (void)n_in; (void)out_size;

  // 1) LN1 -> bf16 activations (async global->LDS staging)
  ln1_kernel<<<MTOT, 256, 0, stream>>>(x, ln1g, ln1b, h_bf);

  // 2) weight transposes / conversions (bf16, [N][K] for WMMA B-fragments)
  {
    dim3 g((NHEAD * HDIM * DMODEL + 255) / 256, 3);
    qkvw_transpose_kernel<<<g, 256, 0, stream>>>(Wq, Wk, Wv, WqT, WkT, WvT);
  }
  transpose_bf16_kernel<<<((size_t)FF * DMODEL + 255) / 256, 256, 0, stream>>>(
      W1, W1T, DMODEL, FF);     // W1T[n*1024+k] = W1[k*4096+n]
  transpose_bf16_kernel<<<((size_t)DMODEL * FF + 255) / 256, 256, 0, stream>>>(
      W2, W2T, FF, DMODEL);     // W2T[n*4096+k] = W2[k*1024+n]

  // 3) QKV projections (WMMA bf16), q pre-scaled, V transposed
  qkv_gemm_kernel<<<1536, 256, 0, stream>>>(h_bf, WqT, WkT, WvT, bq, bk, bv,
                                            q_ws, k_ws, vT_ws);

  // 4) attention (196KB dynamic LDS: scores f32 + probs bf16 + partials)
  int smem = 131072 + 65536 + 4096;
  hipFuncSetAttribute((const void*)attention_kernel,
                      hipFuncAttributeMaxDynamicSharedMemorySize, smem);
  attention_kernel<<<NHEAD * BATCH * (SEQ / 16), 256, smem, stream>>>(
      q_ws, k_ws, vT_ws, attn_p, ent_p, aout);

  // 5) residual + LN2 (async staging)
  addln2_kernel<<<MTOT, 256, 0, stream>>>(x, aout, ln2g, ln2b, y_ws, h2_bf);

  // 6) MLP
  mlp1_kernel<<<2048, 256, 0, stream>>>(h2_bf, W1T, b1, mid);
  mlp2_kernel<<<512, 256, 0, stream>>>(mid, W2T, b2, y_ws, out_p);
}
